// MultiNet_32066225832112
// MI455X (gfx1250) — matrix operand order repaired and checked by reading, not compiled, not run on hardware
//
#include <hip/hip_runtime.h>

#define EPS_BN 1e-5f
#define NPTS 4096
#define KNN_K 20
#define BATCH 2

// ---------------- bf16 helpers (raw ushort storage, RNE convert) ----------------
__device__ __forceinline__ unsigned short f2bf(float f) {
  union { float f; unsigned int u; } v; v.f = f;
  unsigned int r = (v.u + 0x7FFFu + ((v.u >> 16) & 1u)) >> 16;
  return (unsigned short)r;
}
__device__ __forceinline__ float bf2f(unsigned short h) {
  union { unsigned int u; float f; } v; v.u = ((unsigned int)h) << 16; return v.f;
}

// ---------------- gfx1250 async global->LDS staging (guarded; falls back to b128) ----
#if defined(__HIP_DEVICE_COMPILE__) && \
    __has_builtin(__builtin_amdgcn_global_load_async_to_lds_b128) && \
    __has_builtin(__builtin_amdgcn_s_wait_asynccnt)
#define HAS_ASYNC_LDS 1
typedef int v4i_ __attribute__((vector_size(16)));
typedef __attribute__((address_space(1))) v4i_ as1_v4i;
typedef __attribute__((address_space(3))) v4i_ as3_v4i;
#else
#define HAS_ASYNC_LDS 0
#endif

// ---------------- WMMA types / fragment assembly (wave32 layouts per CDNA5 ISA) --
typedef __attribute__((ext_vector_type(16))) __bf16 v16bf;
typedef __attribute__((ext_vector_type(8)))  float  v8f;

union FragU { v16bf v; unsigned int u[8]; };

#define BM 128
#define BN 128
#define BK 32

// A tile in LDS: [BM][BK] row-major bf16. 16-bit A frag (16x32):
// lane L: M = L%16 ; VGPR0..3 = K pairs (kb+2j, kb+2j+1), kb = (L<16)?0:8
//         VGPR4..7 = K pairs (16+kb+2j', ...)
__device__ __forceinline__ void load_a_frag(const unsigned short* As, int mBase, FragU& a) {
  int lane = threadIdx.x & 31;
  const unsigned int* p = (const unsigned int*)(As + (size_t)(mBase + (lane & 15)) * BK);
  int kb = (lane < 16) ? 0 : 4; // dword offset = kb_elems/2
  a.u[0] = p[kb + 0]; a.u[1] = p[kb + 1]; a.u[2] = p[kb + 2]; a.u[3] = p[kb + 3];
  a.u[4] = p[8 + kb + 0]; a.u[5] = p[8 + kb + 1]; a.u[6] = p[8 + kb + 2]; a.u[7] = p[8 + kb + 3];
}

// B tile in LDS: [BN][BK] (N-major; per-column K contiguous). B frag (32x16):
// lane L: N = L%16 ; K = kb+0..7, 16+kb+0..7 with kb = (L<16)?0:8 (same as A)
__device__ __forceinline__ void load_b_frag(const unsigned short* Bs, int nBase, FragU& b) {
  int lane = threadIdx.x & 31;
  const unsigned int* p = (const unsigned int*)(Bs + (size_t)(nBase + (lane & 15)) * BK);
  int kl = (lane < 16) ? 0 : 4;
#pragma unroll
  for (int j = 0; j < 8; ++j) b.u[j] = p[kl + j + (j >= 4 ? 4 : 0)];
}

// per-wave 32x64 sub-tile: 2 A-frags x 4 B-frags -> 8 WMMA per k-step
__device__ __forceinline__ void mma_step(const unsigned short* As, const unsigned short* Bs,
                                         int wm, int wn, v8f acc[2][4]) {
  FragU a[2], b[4];
  load_a_frag(As, wm, a[0]);
  load_a_frag(As, wm + 16, a[1]);
#pragma unroll
  for (int j = 0; j < 4; ++j) load_b_frag(Bs, wn + 16 * j, b[j]);
#pragma unroll
  for (int i = 0; i < 2; ++i)
#pragma unroll
    for (int j = 0; j < 4; ++j)
      acc[i][j] = __builtin_amdgcn_wmma_f32_16x16x32_bf16(false, a[i].v, false, b[j].v,
                                                          (short)0, acc[i][j], false, false);
}

// stage a 128x32 bf16 tile: 256 threads, each moves one 16-elem (32B) group as 2x b128.
// 2 async instructions per wave per call; completion tracked (in-order) by ASYNCcnt.
__device__ __forceinline__ void stage_128x32(const unsigned short* __restrict__ g, size_t gStride,
                                             int k0, unsigned short* l) {
  int row = threadIdx.x >> 1, half = threadIdx.x & 1;
  const unsigned short* gp = g + (size_t)row * gStride + k0 + half * 16;
  unsigned short* lp = l + row * BK + half * 16;
#if HAS_ASYNC_LDS
  __builtin_amdgcn_global_load_async_to_lds_b128((as1_v4i*)gp, (as3_v4i*)lp, 0, 0);
  __builtin_amdgcn_global_load_async_to_lds_b128((as1_v4i*)(gp + 8), (as3_v4i*)(lp + 8), 0, 0);
#else
  *(uint4*)lp = *(const uint4*)gp;
  *(uint4*)(lp + 8) = *(const uint4*)(gp + 8);
#endif
}
// wait until at most NOUT async ops remain outstanding (NOUT = just-issued next tile),
// then barrier so every wave's current tile is visible in LDS.
template<int NOUT>
__device__ __forceinline__ void stage_wait_n() {
#if HAS_ASYNC_LDS
  __builtin_amdgcn_s_wait_asynccnt(NOUT);
#endif
  __syncthreads();
}

// ---------------- generic bf16 GEMM: out(MxN) = A(MxKP) * B(NPxKP N-major),
// M mult of 128, KP mult of 32, B zero-padded to NP=ceil128(N). grid=(tilesN, M/128).
// Double-buffered LDS, async-pipelined. ACT: 0 none, 1 relu, 2 lrelu(0.2). ----------
template<int ACT>
__global__ __launch_bounds__(256) void k_gemm(
    const unsigned short* __restrict__ A, const unsigned short* __restrict__ Bmat,
    const float* __restrict__ scale, const float* __restrict__ shift,
    unsigned short* __restrict__ out, int KP, int N) {
  __shared__ alignas(16) unsigned short As[2][BM * BK];
  __shared__ alignas(16) unsigned short Bs[2][BN * BK];
  int tN = blockIdx.x, tM = blockIdx.y;
  int tid = threadIdx.x, lane = tid & 31, wave = tid >> 5;
  int wm = (wave & 3) * 32, wn = (wave >> 2) * 64;
  const unsigned short* Ab = A + (size_t)tM * BM * KP;
  const unsigned short* Bb = Bmat + (size_t)tN * BN * KP;
  v8f acc[2][4] = {};
  int nk = KP / BK;
  stage_128x32(Ab, KP, 0, As[0]);
  stage_128x32(Bb, KP, 0, Bs[0]);
  for (int s = 0; s < nk - 1; ++s) {
    int cur = s & 1, nxt = cur ^ 1;
    stage_128x32(Ab, KP, (s + 1) * BK, As[nxt]);  // prefetch next tile (async)
    stage_128x32(Bb, KP, (s + 1) * BK, Bs[nxt]);
    stage_wait_n<4>();                            // current tile landed; next in flight
    mma_step(As[cur], Bs[cur], wm, wn, acc);
    __syncthreads();                              // done reading cur before overwrite
  }
  stage_wait_n<0>();
  mma_step(As[(nk - 1) & 1], Bs[(nk - 1) & 1], wm, wn, acc);
#pragma unroll
  for (int i = 0; i < 2; ++i)
#pragma unroll
    for (int j = 0; j < 4; ++j) {
      int col = tN * BN + wn + j * 16 + (lane & 15);
      if (col >= N) continue;
      float sc = scale[col], sh = shift[col];
      int rbase = tM * BM + wm + i * 16 + (lane >> 4) * 8;
      unsigned short* op = out + (unsigned)(rbase * N + col); // all outputs < 2^31 elems
#pragma unroll
      for (int r = 0; r < 8; ++r) {
        float v = acc[i][j][r] * sc + sh;
        if (ACT == 1) v = fmaxf(v, 0.f);
        if (ACT == 2) v = (v >= 0.f) ? v : 0.2f * v;
        op[(unsigned)(r * N)] = f2bf(v);
      }
    }
}

// ---------------- implicit-GEMM 3x3x3 conv A-gather (one voxel row, 16-elem k-group) ----
template<int CIL>
__device__ __forceinline__ void conv_gather(const unsigned short* __restrict__ in, int S,
                                            int ar, int kk0, int dq, int hq, int wq, int bq,
                                            int k0, unsigned short* As) {
  constexpr int Ci = 1 << CIL;
  constexpr int K = 27 * Ci;
  if (CIL >= 6) {
    // all 16 k's share one (kd,kh,kw); contiguous Ci-slice -> 2x16B vector copy
    int k = k0 + kk0;
    int t = k >> CIL, ci0 = k & (Ci - 1);
    int kd = t / 9, kh = (t / 3) % 3, kw = t % 3;
    int id = dq + kd - 1, ih = hq + kh - 1, iw = wq + kw - 1;
    uint4 v0 = {0, 0, 0, 0}, v1 = {0, 0, 0, 0};
    if (id >= 0 && id < S && ih >= 0 && ih < S && iw >= 0 && iw < S) {
      const uint4* src = (const uint4*)&in[((((size_t)bq * S + id) * S + ih) * S + iw) * Ci + ci0];
      v0 = src[0]; v1 = src[1];
    }
    *(uint4*)&As[ar * BK + kk0] = v0;
    *(uint4*)&As[ar * BK + kk0 + 8] = v1;
  } else {
#pragma unroll
    for (int q = 0; q < 16; ++q) {
      int k = k0 + kk0 + q;
      unsigned short v = 0;
      if (k < K) {
        int t = k >> CIL, ci = k & (Ci - 1);
        int kd = t / 9, kh = (t / 3) % 3, kw = t % 3;
        int id = dq + kd - 1, ih = hq + kh - 1, iw = wq + kw - 1;
        if (id >= 0 && id < S && ih >= 0 && ih < S && iw >= 0 && iw < S)
          v = in[((((size_t)bq * S + id) * S + ih) * S + iw) * Ci + ci];
      }
      As[ar * BK + kk0 + q] = v;
    }
  }
}

// implicit-GEMM conv, pad 1, stride 1, NDHWC bf16; weights padded NPxKP, async-pipelined.
template<int ACT, int CIL>
__global__ __launch_bounds__(256) void k_conv_gemm(
    const unsigned short* __restrict__ in, const unsigned short* __restrict__ Wm,
    const float* __restrict__ scale, const float* __restrict__ shift,
    unsigned short* __restrict__ out, int S, int Co) {
  constexpr int Ci = 1 << CIL;
  constexpr int K = 27 * Ci;
  constexpr int KP = (K + 31) & ~31;
  constexpr int NK = KP / BK;
  __shared__ alignas(16) unsigned short As[2][BM * BK];
  __shared__ alignas(16) unsigned short Bs[2][BN * BK];
  int tN = blockIdx.x, tM = blockIdx.y;
  int tid = threadIdx.x, lane = tid & 31, wave = tid >> 5;
  int wm = (wave & 3) * 32, wn = (wave >> 2) * 64;
  int ar = tid >> 1, kk0 = (tid & 1) * 16;
  int m = tM * BM + ar; // always < M by construction
  int wq = m % S, hq = (m / S) % S, dq = (m / (S * S)) % S, bq = m / (S * S * S);
  const unsigned short* Bb = Wm + (size_t)tN * BN * KP;
  v8f acc[2][4] = {};
  conv_gather<CIL>(in, S, ar, kk0, dq, hq, wq, bq, 0, As[0]);
  stage_128x32(Bb, KP, 0, Bs[0]);
  for (int s = 0; s < NK - 1; ++s) {
    int cur = s & 1, nxt = cur ^ 1;
    conv_gather<CIL>(in, S, ar, kk0, dq, hq, wq, bq, (s + 1) * BK, As[nxt]);
    stage_128x32(Bb, KP, (s + 1) * BK, Bs[nxt]);  // async prefetch of next weight panel
    stage_wait_n<2>();
    mma_step(As[cur], Bs[cur], wm, wn, acc);
    __syncthreads();
  }
  stage_wait_n<0>();
  mma_step(As[(NK - 1) & 1], Bs[(NK - 1) & 1], wm, wn, acc);
#pragma unroll
  for (int i = 0; i < 2; ++i)
#pragma unroll
    for (int j = 0; j < 4; ++j) {
      int col = tN * BN + wn + j * 16 + (lane & 15);
      if (col >= Co) continue;
      float sc = scale[col], sh = shift[col];
      int rbase = tM * BM + wm + i * 16 + (lane >> 4) * 8;
      unsigned short* op = out + (unsigned)(rbase * Co + col);
#pragma unroll
      for (int r = 0; r < 8; ++r) {
        float v = acc[i][j][r] * sc + sh;
        if (ACT == 1) v = fmaxf(v, 0.f);
        if (ACT == 2) v = (v >= 0.f) ? v : 0.2f * v;
        op[(unsigned)(r * Co)] = f2bf(v);
      }
    }
}

// ---------------- 2x2x2 max-pool on NDHWC bf16 ----------------
__global__ void k_pool(const unsigned short* __restrict__ in, unsigned short* __restrict__ out,
                       int Bn, int S, int C) {
  int S2 = S >> 1;
  size_t total = (size_t)Bn * S2 * S2 * S2 * C;
  size_t e = (size_t)blockIdx.x * blockDim.x + threadIdx.x;
  if (e >= total) return;
  int c = (int)(e % C); size_t r = e / C;
  int w2 = (int)(r % S2); r /= S2;
  int h2 = (int)(r % S2); r /= S2;
  int d2 = (int)(r % S2); int b = (int)(r / S2);
  float mx = -3.4e38f;
#pragma unroll
  for (int dz = 0; dz < 2; ++dz)
#pragma unroll
    for (int dy = 0; dy < 2; ++dy)
#pragma unroll
      for (int dx = 0; dx < 2; ++dx) {
        size_t idx = ((((size_t)b * S + (2 * d2 + dz)) * S + (2 * h2 + dy)) * S + (2 * w2 + dx)) * C + c;
        mx = fmaxf(mx, bf2f(in[idx]));
      }
  out[e] = f2bf(mx);
}

// ---------------- weight/param prep (zero-padded NPxKP, N-major) ----------------
__global__ void k_prep_convw(const float* __restrict__ w, unsigned short* __restrict__ out,
                             int Ci, int Co, int KP, int NP) {
  // (Co,Ci,27) f32 -> Bmat[n][k], k = t*Ci + ci
  size_t total = (size_t)NP * KP;
  size_t e = (size_t)blockIdx.x * blockDim.x + threadIdx.x;
  if (e >= total) return;
  int k = (int)(e % KP); int n = (int)(e / KP);
  int K = 27 * Ci;
  unsigned short v = 0;
  if (n < Co && k < K) {
    int t = k / Ci, ci = k % Ci;
    v = f2bf(w[((size_t)n * Ci + ci) * 27 + t]);
  }
  out[e] = v;
}
__global__ void k_prep_pad(const float* __restrict__ w, unsigned short* __restrict__ out,
                           int Cin, int Co, int KP, int NP) {
  // (Co,Cin) f32 -> Bmat[n][k] = w[n][k], zero-padded
  size_t total = (size_t)NP * KP;
  size_t e = (size_t)blockIdx.x * blockDim.x + threadIdx.x;
  if (e >= total) return;
  int k = (int)(e % KP); int n = (int)(e / KP);
  out[e] = (n < Co && k < Cin) ? f2bf(w[(size_t)n * Cin + k]) : (unsigned short)0;
}
__global__ void k_fold(const float* __restrict__ g, const float* __restrict__ be,
                       const float* __restrict__ cb, float* scale, float* shift, int Co) {
  int i = blockIdx.x * blockDim.x + threadIdx.x;
  if (i >= Co) return;
  float s = g[i] * rsqrtf(1.f + EPS_BN);
  scale[i] = s;
  shift[i] = (cb ? cb[i] * s : 0.f) + be[i];
}
__global__ void k_f32_to_bf16(const float* __restrict__ x, unsigned short* __restrict__ o, size_t n) {
  size_t e = (size_t)blockIdx.x * blockDim.x + threadIdx.x;
  if (e < n) o[e] = f2bf(x[e]);
}
__global__ void k_x_to_bnc(const float* __restrict__ x, unsigned short* __restrict__ o) {
  // (B,3,N) f32 -> (B,N,3) bf16
  size_t total = (size_t)BATCH * NPTS * 3;
  size_t e = (size_t)blockIdx.x * blockDim.x + threadIdx.x;
  if (e >= total) return;
  int c = (int)(e % 3); size_t r = e / 3;
  int n = (int)(r % NPTS); int b = (int)(r / NPTS);
  o[e] = f2bf(x[((size_t)b * 3 + c) * NPTS + n]);
}

// ---------------- kNN: per point, distance row in LDS + 20 argmin passes ----------------
__global__ __launch_bounds__(256) void k_knn(const unsigned short* __restrict__ act, int stride,
                                             int C, int* __restrict__ idxOut) {
  int i = blockIdx.x % NPTS, b = blockIdx.x / NPTS;
  __shared__ float sxi[128];
  __shared__ float sd[NPTS];
  __shared__ float sval[256];
  __shared__ int   sidx[256];
  const unsigned short* base = act + (size_t)b * NPTS * stride;
  for (int c = threadIdx.x; c < C; c += 256) sxi[c] = bf2f(base[(size_t)i * stride + c]);
  __syncthreads();
  for (int j = threadIdx.x; j < NPTS; j += 256) {
    const unsigned short* pj = base + (size_t)j * stride;
    float s = 0.f;
    for (int c = 0; c < C; ++c) { float d = sxi[c] - bf2f(pj[c]); s += d * d; }
    sd[j] = s;
  }
  __syncthreads();
  for (int t = 0; t < KNN_K; ++t) {
    float bv = 3.4e38f; int bi = NPTS;
    for (int j = threadIdx.x; j < NPTS; j += 256)
      if (sd[j] < bv) { bv = sd[j]; bi = j; }
    sval[threadIdx.x] = bv; sidx[threadIdx.x] = bi;
    __syncthreads();
    for (int st = 128; st > 0; st >>= 1) {
      if (threadIdx.x < st) {
        if (sval[threadIdx.x + st] < sval[threadIdx.x] ||
            (sval[threadIdx.x + st] == sval[threadIdx.x] && sidx[threadIdx.x + st] < sidx[threadIdx.x])) {
          sval[threadIdx.x] = sval[threadIdx.x + st];
          sidx[threadIdx.x] = sidx[threadIdx.x + st];
        }
      }
      __syncthreads();
    }
    if (threadIdx.x == 0) {
      idxOut[((size_t)b * NPTS + i) * KNN_K + t] = sidx[0];
      sd[sidx[0]] = 3.4e38f;
    }
    __syncthreads();
  }
}

// ---------------- edge feature: F[(b,n,kk), 0:KP] = cat(nbr-ctr, ctr, 0-pad) -------------
__global__ void k_edgefeat(const unsigned short* __restrict__ act, int stride, int C,
                           const int* __restrict__ idx, unsigned short* __restrict__ F, int KP) {
  int twoC = 2 * C;
  size_t total = (size_t)BATCH * NPTS * KNN_K * KP;
  size_t e = (size_t)blockIdx.x * blockDim.x + threadIdx.x;
  if (e >= total) return;
  int col = (int)(e % KP); size_t r = e / KP;
  int kk = (int)(r % KNN_K); size_t bn = r / KNN_K; // bn = b*NPTS + n
  int b = (int)(bn / NPTS);
  float v = 0.f;
  if (col < twoC) {
    int j = idx[bn * KNN_K + kk];
    if (col < C)
      v = bf2f(act[((size_t)b * NPTS + j) * stride + col]) - bf2f(act[bn * stride + col]);
    else
      v = bf2f(act[bn * stride + (col - C)]);
  }
  F[r * KP + col] = f2bf(v);
}

// ---------------- max over k=20, write into column slice of xc (B,N,512) ----------------
__global__ void k_maxk(const unsigned short* __restrict__ z, int Co,
                       unsigned short* __restrict__ xc, int colOff) {
  size_t total = (size_t)BATCH * NPTS * Co;
  size_t e = (size_t)blockIdx.x * blockDim.x + threadIdx.x;
  if (e >= total) return;
  int o = (int)(e % Co); size_t bn = e / Co;
  float mx = -3.4e38f;
  for (int kk = 0; kk < KNN_K; ++kk)
    mx = fmaxf(mx, bf2f(z[(bn * KNN_K + kk) * (size_t)Co + o]));
  xc[bn * 512 + colOff + o] = f2bf(mx);
}

// ---------------- max & mean over points of z5 (B,N,1024) -> feat[:,0:2048] ----------------
__global__ __launch_bounds__(256) void k_maxmean(const unsigned short* __restrict__ z5,
                                                 float* __restrict__ feat) {
  int ch = blockIdx.x % 1024, b = blockIdx.x / 1024;
  __shared__ float smax[256], ssum[256];
  float mx = -3.4e38f, sm = 0.f;
  for (int n = threadIdx.x; n < NPTS; n += 256) {
    float v = bf2f(z5[((size_t)b * NPTS + n) * 1024 + ch]);
    mx = fmaxf(mx, v); sm += v;
  }
  smax[threadIdx.x] = mx; ssum[threadIdx.x] = sm;
  __syncthreads();
  for (int st = 128; st > 0; st >>= 1) {
    if (threadIdx.x < st) {
      smax[threadIdx.x] = fmaxf(smax[threadIdx.x], smax[threadIdx.x + st]);
      ssum[threadIdx.x] += ssum[threadIdx.x + st];
    }
    __syncthreads();
  }
  if (threadIdx.x == 0) {
    feat[(size_t)b * 4096 + ch] = smax[0];
    feat[(size_t)b * 4096 + 1024 + ch] = ssum[0] / (float)NPTS;
  }
}

// ---------------- fc1: NDHWC(B,4,4,4,512) flattened as NCDHW, relu ----------------
__global__ __launch_bounds__(256) void k_fc1(const unsigned short* __restrict__ act,
                                             const float* __restrict__ w, const float* __restrict__ bias,
                                             float* __restrict__ feat) {
  int o = blockIdx.x % 2048, b = blockIdx.x / 2048;
  __shared__ float red[256];
  float s = 0.f;
  for (int t = threadIdx.x; t < 32768; t += 256) {
    int c = t >> 6, sp = t & 63; // NCDHW flat t = c*64 + sp
    __builtin_prefetch(&w[(size_t)o * 32768 + t + 2048], 0, 0);
    s += bf2f(act[((size_t)b * 64 + sp) * 512 + c]) * w[(size_t)o * 32768 + t];
  }
  red[threadIdx.x] = s; __syncthreads();
  for (int st = 128; st > 0; st >>= 1) {
    if (threadIdx.x < st) red[threadIdx.x] += red[threadIdx.x + st];
    __syncthreads();
  }
  if (threadIdx.x == 0)
    feat[(size_t)b * 4096 + 2048 + o] = fmaxf(red[0] + bias[o], 0.f);
}

// ---------------- generic small FC (head): dot + bias? + bn? + act ----------------
template<int ACT>
__global__ __launch_bounds__(256) void k_fc(const float* __restrict__ in, int inStride,
                                            const float* __restrict__ w, const float* __restrict__ bias,
                                            const float* __restrict__ gamma, const float* __restrict__ beta,
                                            float* __restrict__ out, int outStride, int In, int Out) {
  int o = blockIdx.x % Out, b = blockIdx.x / Out;
  __shared__ float red[256];
  float s = 0.f;
  for (int t = threadIdx.x; t < In; t += 256) {
    __builtin_prefetch(&w[(size_t)o * In + t + 1024], 0, 0);
    s += in[(size_t)b * inStride + t] * w[(size_t)o * In + t];
  }
  red[threadIdx.x] = s; __syncthreads();
  for (int st = 128; st > 0; st >>= 1) {
    if (threadIdx.x < st) red[threadIdx.x] += red[threadIdx.x + st];
    __syncthreads();
  }
  if (threadIdx.x == 0) {
    float v = red[0];
    if (bias) v += bias[o];
    if (gamma) v = v * (gamma[o] * rsqrtf(1.f + EPS_BN)) + beta[o];
    if (ACT == 2) v = (v >= 0.f) ? v : 0.2f * v;
    out[(size_t)b * outStride + o] = v;
  }
}

// =======================================================================================
extern "C" void kernel_launch(void* const* d_in, const int* in_sizes, int n_in,
                              void* d_out, int out_size, void* d_ws, size_t ws_size,
                              hipStream_t stream) {
  (void)in_sizes; (void)n_in; (void)out_size; (void)ws_size;
  const float* x_in = (const float*)d_in[0];
  const float* y_in = (const float*)d_in[1];

  auto gdiv = [](size_t a, size_t b) -> unsigned int { return (unsigned int)((a + b - 1) / b); };
  auto padK = [](int k) -> int { return (k + 31) & ~31; };
  auto padN = [](int n) -> int { return (n + 127) & ~127; };

  // ---- workspace bump allocator ----
  char* wsp = (char*)d_ws; size_t off = 0;
  auto carve = [&](size_t bytes) -> void* {
    void* p = wsp + off; off += (bytes + 255) & ~(size_t)255; return p;
  };

  // conv layers: {wIdx(b=+1,g=+2,be=+3), Ci, Co, S_in, poolAfter}
  struct CL { int wi, Ci, Co, S, pool; };
  const CL cls[8] = {
    { 2,   1,  64, 128, 1}, { 6,  64, 128, 64, 1}, {10, 128, 256, 32, 0}, {14, 256, 256, 32, 1},
    {18, 256, 512,  16, 0}, {22, 512, 512, 16, 1}, {26, 512, 512,  8, 0}, {30, 512, 512,  8, 1}
  };
  unsigned short* convW[8]; float* cScale[8]; float* cShift[8];
  for (int l = 0; l < 8; ++l) {
    size_t ke = (size_t)padN(cls[l].Co) * padK(27 * cls[l].Ci);
    convW[l]  = (unsigned short*)carve(ke * 2);
    cScale[l] = (float*)carve((size_t)cls[l].Co * 4);
    cShift[l] = (float*)carve((size_t)cls[l].Co * 4);
  }
  // edge layers: {wIdx(g=+1,be=+2), C(branch), Co, colOff}
  struct EL { int wi, C, Co, colOff; };
  const EL els[4] = { {36, 3, 64, 0}, {39, 64, 64, 64}, {42, 64, 128, 128}, {45, 128, 256, 256} };
  unsigned short* edgeW[4]; float* eScale[4]; float* eShift[4];
  for (int l = 0; l < 4; ++l) {
    edgeW[l]  = (unsigned short*)carve((size_t)padN(els[l].Co) * padK(2 * els[l].C) * 2);
    eScale[l] = (float*)carve((size_t)els[l].Co * 4);
    eShift[l] = (float*)carve((size_t)els[l].Co * 4);
  }
  unsigned short* edgeW5 = (unsigned short*)carve((size_t)1024 * 512 * 2);
  float* e5Scale = (float*)carve(1024 * 4);
  float* e5Shift = (float*)carve(1024 * 4);

  // big ping-pong activation buffers (NDHWC bf16)
  unsigned short* bufA = (unsigned short*)carve((size_t)BATCH * 128 * 128 * 128 * 64 * 2); // 536 MB
  unsigned short* bufB = (unsigned short*)carve((size_t)BATCH * 64 * 64 * 64 * 128 * 2);   // 134 MB

  // DGCNN scratch aliases bufA (conv branch is done with it before DGCNN starts)
  char* p2 = (char*)bufA; size_t off2 = 0;
  auto carve2 = [&](size_t bytes) -> void* {
    void* p = p2 + off2; off2 += (bytes + 255) & ~(size_t)255; return p;
  };
  int*            idxBuf = (int*)carve2((size_t)BATCH * NPTS * KNN_K * 4);
  unsigned short* xt0    = (unsigned short*)carve2((size_t)BATCH * NPTS * 3 * 2);
  unsigned short* Fbuf   = (unsigned short*)carve2((size_t)BATCH * NPTS * KNN_K * 512 * 2); // KP<=512
  unsigned short* zBuf   = (unsigned short*)carve2((size_t)BATCH * NPTS * KNN_K * 256 * 2); // Co<=256
  unsigned short* xcBuf  = (unsigned short*)carve2((size_t)BATCH * NPTS * 512 * 2);
  unsigned short* z5Buf  = (unsigned short*)carve2((size_t)BATCH * NPTS * 1024 * 2);
  float* o1 = (float*)carve2((size_t)BATCH * 2048 * 4);
  float* o2 = (float*)carve2((size_t)BATCH * 1024 * 4);
  float* o3 = (float*)carve2((size_t)BATCH * 512 * 4);

  float* feat = (float*)d_out + 4; // (B,4096): [max 1024 | mean 1024 | img 2048]

  // ---- param prep ----
  for (int l = 0; l < 8; ++l) {
    int KP = padK(27 * cls[l].Ci), NP = padN(cls[l].Co);
    size_t ke = (size_t)NP * KP;
    k_prep_convw<<<gdiv(ke, 256), 256, 0, stream>>>((const float*)d_in[cls[l].wi], convW[l],
                                                    cls[l].Ci, cls[l].Co, KP, NP);
    k_fold<<<gdiv(cls[l].Co, 256), 256, 0, stream>>>((const float*)d_in[cls[l].wi + 2],
                                                     (const float*)d_in[cls[l].wi + 3],
                                                     (const float*)d_in[cls[l].wi + 1],
                                                     cScale[l], cShift[l], cls[l].Co);
  }
  for (int l = 0; l < 4; ++l) {
    int KP = padK(2 * els[l].C), NP = padN(els[l].Co);
    size_t ke = (size_t)NP * KP;
    k_prep_pad<<<gdiv(ke, 256), 256, 0, stream>>>((const float*)d_in[els[l].wi], edgeW[l],
                                                  2 * els[l].C, els[l].Co, KP, NP);
    k_fold<<<gdiv(els[l].Co, 256), 256, 0, stream>>>((const float*)d_in[els[l].wi + 1],
                                                     (const float*)d_in[els[l].wi + 2],
                                                     nullptr, eScale[l], eShift[l], els[l].Co);
  }
  k_prep_pad<<<gdiv((size_t)1024 * 512, 256), 256, 0, stream>>>((const float*)d_in[48], edgeW5,
                                                                512, 1024, 512, 1024);
  k_fold<<<gdiv(1024, 256), 256, 0, stream>>>((const float*)d_in[49], (const float*)d_in[50],
                                              nullptr, e5Scale, e5Shift, 1024);

  // ---- voxel CNN branch (ping-pong: bufB <-> bufA) ----
  size_t yN = (size_t)BATCH * 128 * 128 * 128; // C=1 -> flat NDHWC == NCDHW
  k_f32_to_bf16<<<gdiv(yN, 256), 256, 0, stream>>>(y_in, bufB, yN);
  unsigned short* pa = bufB;
  unsigned short* pb = bufA;
  for (int l = 0; l < 8; ++l) {
    int S = cls[l].S, Ci = cls[l].Ci, Co = cls[l].Co;
    size_t M = (size_t)BATCH * S * S * S; // multiple of 128 for all layers
    dim3 grid(gdiv(Co, BN), (unsigned int)(M / BM));
    switch (Ci) {
      case 1:   k_conv_gemm<1, 0><<<grid, 256, 0, stream>>>(pa, convW[l], cScale[l], cShift[l], pb, S, Co); break;
      case 64:  k_conv_gemm<1, 6><<<grid, 256, 0, stream>>>(pa, convW[l], cScale[l], cShift[l], pb, S, Co); break;
      case 128: k_conv_gemm<1, 7><<<grid, 256, 0, stream>>>(pa, convW[l], cScale[l], cShift[l], pb, S, Co); break;
      case 256: k_conv_gemm<1, 8><<<grid, 256, 0, stream>>>(pa, convW[l], cScale[l], cShift[l], pb, S, Co); break;
      default:  k_conv_gemm<1, 9><<<grid, 256, 0, stream>>>(pa, convW[l], cScale[l], cShift[l], pb, S, Co); break;
    }
    { unsigned short* t = pa; pa = pb; pb = t; }
    if (cls[l].pool) {
      int S2 = S >> 1;
      size_t tot = (size_t)BATCH * S2 * S2 * S2 * Co;
      k_pool<<<gdiv(tot, 256), 256, 0, stream>>>(pa, pb, BATCH, S, Co);
      { unsigned short* t = pa; pa = pb; pb = t; }
    }
  }
  // final pooled activation (B,4,4,4,512) in pa (== bufA); fc1 -> feat img part (must run
  // BEFORE DGCNN scratch, which aliases bufA, is written)
  k_fc1<<<BATCH * 2048, 256, 0, stream>>>(pa, (const float*)d_in[34], (const float*)d_in[35], feat);

  // ---- DGCNN branch ----
  k_x_to_bnc<<<gdiv((size_t)BATCH * NPTS * 3, 256), 256, 0, stream>>>(x_in, xt0);
  for (int l = 0; l < 4; ++l) {
    int C = els[l].C, Co = els[l].Co;
    int KP = padK(2 * C);
    const unsigned short* src; int stride;
    if (l == 0)      { src = xt0;         stride = 3;   }
    else if (l == 1) { src = xcBuf + 0;   stride = 512; }
    else if (l == 2) { src = xcBuf + 64;  stride = 512; }
    else             { src = xcBuf + 128; stride = 512; }
    k_knn<<<BATCH * NPTS, 256, 0, stream>>>(src, stride, C, idxBuf);
    size_t totF = (size_t)BATCH * NPTS * KNN_K * KP;
    k_edgefeat<<<gdiv(totF, 256), 256, 0, stream>>>(src, stride, C, idxBuf, Fbuf, KP);
    size_t M = (size_t)BATCH * NPTS * KNN_K; // 163840, multiple of 128
    dim3 grid(gdiv(Co, BN), (unsigned int)(M / BM));
    k_gemm<2><<<grid, 256, 0, stream>>>(Fbuf, edgeW[l], eScale[l], eShift[l], zBuf, KP, Co);
    size_t totZ = (size_t)BATCH * NPTS * Co;
    k_maxk<<<gdiv(totZ, 256), 256, 0, stream>>>(zBuf, Co, xcBuf, els[l].colOff);
  }
  // c5: (B*N, 512) x (512 -> 1024), bn+lrelu
  {
    size_t M = (size_t)BATCH * NPTS; // 8192
    dim3 grid(gdiv(1024, BN), (unsigned int)(M / BM));
    k_gemm<2><<<grid, 256, 0, stream>>>(xcBuf, edgeW5, e5Scale, e5Shift, z5Buf, 512, 1024);
  }
  k_maxmean<<<BATCH * 1024, 256, 0, stream>>>(z5Buf, feat);

  // ---- MLP head ----
  k_fc<2><<<BATCH * 2048, 256, 0, stream>>>(feat, 4096, (const float*)d_in[51], nullptr,
                                            (const float*)d_in[52], (const float*)d_in[53],
                                            o1, 2048, 4096, 2048);
  k_fc<2><<<BATCH * 1024, 256, 0, stream>>>(o1, 2048, (const float*)d_in[54], (const float*)d_in[55],
                                            (const float*)d_in[56], (const float*)d_in[57],
                                            o2, 1024, 2048, 1024);
  k_fc<2><<<BATCH * 512, 256, 0, stream>>>(o2, 1024, (const float*)d_in[58], (const float*)d_in[59],
                                           (const float*)d_in[60], (const float*)d_in[61],
                                           o3, 512, 1024, 512);
  k_fc<0><<<BATCH * 2, 256, 0, stream>>>(o3, 512, (const float*)d_in[62], (const float*)d_in[63],
                                         nullptr, nullptr, (float*)d_out, 2, 512, 2);
}
